// BatchFunctionalVGG_63273458205361
// MI455X (gfx1250) — compile-verified, matrix operand order checked by reading, not run on hardware
//
#include <hip/hip_runtime.h>

typedef __attribute__((ext_vector_type(16))) _Float16 v16h;
typedef __attribute__((ext_vector_type(8)))  _Float16 v8h;
typedef __attribute__((ext_vector_type(8)))  float    v8f;

#define MODEL_BS 4
#define IMG_BS   32

// ===========================================================================
// Packing kernels (bandwidth-trivial vs the 80 GFLOP of matrix math)
// ===========================================================================

// f32 NCHW [32,3,32,32] -> f16 NHWC [32,32,32,32] (channels zero-padded to 32)
__global__ void pack_input(const float* __restrict__ x, _Float16* __restrict__ o,
                           int total)
{
    int i = blockIdx.x * blockDim.x + threadIdx.x;
    if (i >= total) return;
    int c   = i & 31;
    int t2  = i >> 5;
    int hw  = t2 & 1023;
    int img = t2 >> 10;
    o[i] = (c < 3) ? (_Float16)x[(img * 3 + c) * 1024 + hw] : (_Float16)0.f;
}

// conv w f32 [b][ci][co][KK] -> f16 [b][KK][co][Cpad] (ci contiguous, padded)
// grid: x covers co*ci (Cpad power of two), y = tap, z = model
__global__ void pack_weights(const float* __restrict__ w, _Float16* __restrict__ o,
                             int Cin, int cpadShift, int Co, int KK)
{
    int idx = blockIdx.x * blockDim.x + threadIdx.x;
    int ci  = idx & ((1 << cpadShift) - 1);
    int co  = idx >> cpadShift;
    if (co >= Co) return;
    int t = blockIdx.y, b = blockIdx.z;
    _Float16 v = (_Float16)0.f;
    if (ci < Cin)
        v = (_Float16)w[((b * Cin + ci) * Co + co) * KK + t];
    o[(((b * KK + t) * Co + co) << cpadShift) + ci] = v;
}

// ===========================================================================
// Tap-decomposed implicit-GEMM 3x3 SAME conv + bias + ReLU, f16 NHWC.
// GEMM: M = Cout, N = IMG_BS*H*W, K = 9 taps x Cinp (chunks of 32 channels).
// Block = 256 threads (8 waves), tile = 64(M) x 128(N).
// Wave w owns M-tiles {2*(w>>2), +1} x N-tiles {2*(w&3), +1} -> 4 WMMA/K-step.
// All offsets are 32-bit; H, W, HW are powers of two (shift-decomposed).
// ===========================================================================
__global__ __launch_bounds__(256)
void conv3x3_wmma(const _Float16* __restrict__ x, const _Float16* __restrict__ wp,
                  const float* __restrict__ bias, _Float16* __restrict__ y,
                  int Cinp, int Cout, int H, int W, int wShift,
                  int in_model_stride)
{
    const int b       = blockIdx.z;
    const int co_base = blockIdx.y * 64;
    const int n0      = blockIdx.x * 128;
    const int HW      = H * W;
    const int tid     = threadIdx.x;
    const int lane    = tid & 31;
    const int wave    = tid >> 5;

    __shared__ alignas(16) _Float16 As[64 * 32];    // [m][cc]
    __shared__ alignas(16) _Float16 Bs[128 * 32];   // [n][cc]

    const _Float16* xin = x  + b * in_model_stride;
    const _Float16* wb  = wp + b * 9 * Cinp * Cout;

    // wave -> tile mapping
    const int nt0 = (wave & 3) * 2;
    const int mt0 = (wave >> 2) * 2;

    // fragment read pattern (wave32 16-bit layout)
    const int frow = lane & 15;
    const int kb   = (lane < 16) ? 0 : 8;

    // A staging: rowA 0..63, ccA in {0,8,16,24}
    const int rowA    = tid >> 2;
    const int ccA     = (tid & 3) * 8;
    const int rowAOff = (co_base + rowA) * Cinp + ccA;
    // B staging: rowB 0..127, ccB in {0,16}
    const int rowB = tid >> 1;
    const int ccB  = (tid & 1) * 16;

    // B spatial decomposition -- once per thread, shift/mask only
    const int nS      = n0 + rowB;
    const int rem     = nS & (HW - 1);          // position within image
    const int imgBase = (nS - rem) * Cinp;      // img * HW * Cinp
    const int h       = rem >> wShift;
    const int wcol    = rem & (W - 1);

    v8f acc00 = {}, acc01 = {}, acc10 = {}, acc11 = {};

    const int cchunks = Cinp >> 5;
    for (int t = 0; t < 9; ++t) {
        const int  dh = t / 3 - 1;               // t is uniform: SALU only
        const int  dw = t - (t / 3) * 3 - 1;
        const int  hi = h + dh, wi = wcol + dw;
        const bool vspat = (hi >= 0) && (hi < H) && (wi >= 0) && (wi < W);
        // 32-bit offset; invalid taps clamp to image base (zero-selected later)
        const int spOff  = imgBase + (vspat ? (rem + dh * W + dw) * Cinp : 0);
        const int wtBase = t * Cout * Cinp;

        __builtin_prefetch(&wb[wtBase + rowAOff], 0, 1);

        for (int c0 = 0; c0 < cchunks * 32; c0 += 32) {
            // ---- stage B: 2 contiguous v8h loads + 2 ds_store_b128 --------
            {
                v8h r0 = *(const v8h*)&xin[spOff + c0 + ccB];
                v8h r1 = *(const v8h*)&xin[spOff + c0 + ccB + 8];
                v8h z  = {};
                v8h p0 = vspat ? r0 : z;
                v8h p1 = vspat ? r1 : z;
                *(v8h*)&Bs[rowB * 32 + ccB]     = p0;
                *(v8h*)&Bs[rowB * 32 + ccB + 8] = p1;
            }
            // ---- stage A: 1 contiguous v8h load + 1 ds_store_b128 ---------
            *(v8h*)&As[rowA * 32 + ccA] = *(const v8h*)&wb[wtBase + rowAOff + c0];

            __syncthreads();

            // ---- fragments + 4 WMMA per wave ------------------------------
            const v8h a00 = *(const v8h*)&As[(mt0 * 16 + frow) * 32 + kb];
            const v8h a01 = *(const v8h*)&As[(mt0 * 16 + frow) * 32 + kb + 16];
            const v8h a10 = *(const v8h*)&As[((mt0 + 1) * 16 + frow) * 32 + kb];
            const v8h a11 = *(const v8h*)&As[((mt0 + 1) * 16 + frow) * 32 + kb + 16];
            const v8h b00 = *(const v8h*)&Bs[(nt0 * 16 + frow) * 32 + kb];
            const v8h b01 = *(const v8h*)&Bs[(nt0 * 16 + frow) * 32 + kb + 16];
            const v8h b10 = *(const v8h*)&Bs[((nt0 + 1) * 16 + frow) * 32 + kb];
            const v8h b11 = *(const v8h*)&Bs[((nt0 + 1) * 16 + frow) * 32 + kb + 16];
            v16h av0, av1, bv0, bv1;
            #pragma unroll
            for (int i = 0; i < 8; ++i) {
                av0[i] = a00[i]; av0[8 + i] = a01[i];
                av1[i] = a10[i]; av1[8 + i] = a11[i];
                bv0[i] = b00[i]; bv0[8 + i] = b01[i];
                bv1[i] = b10[i]; bv1[8 + i] = b11[i];
            }
            acc00 = __builtin_amdgcn_wmma_f32_16x16x32_f16(
                        false, av0, false, bv0, (short)0, acc00, false, false);
            acc01 = __builtin_amdgcn_wmma_f32_16x16x32_f16(
                        false, av0, false, bv1, (short)0, acc01, false, false);
            acc10 = __builtin_amdgcn_wmma_f32_16x16x32_f16(
                        false, av1, false, bv0, (short)0, acc10, false, false);
            acc11 = __builtin_amdgcn_wmma_f32_16x16x32_f16(
                        false, av1, false, bv1, (short)0, acc11, false, false);
            __syncthreads();
        }
    }

    // ---- epilogue: bias + ReLU, contiguous v8h stores (NHWC: no div) ------
    const int mofs = (lane < 16) ? 0 : 8;
    const int co0  = co_base + mt0 * 16 + mofs;        // 8 contiguous channels
    const int co1  = co0 + 16;
    const v8f bias0 = *(const v8f*)&bias[b * Cout + co0];
    const v8f bias1 = *(const v8f*)&bias[b * Cout + co1];
    const int yb = b * (IMG_BS * HW * Cout);

    #pragma unroll
    for (int ni = 0; ni < 2; ++ni) {
        const int n_col = n0 + (nt0 + ni) * 16 + frow;
        const int ob    = yb + n_col * Cout;
        const v8f& a0 = ni ? acc01 : acc00;
        const v8f& a1 = ni ? acc11 : acc10;
        v8h o0, o1;
        #pragma unroll
        for (int r = 0; r < 8; ++r) {
            o0[r] = (_Float16)fmaxf(a0[r] + bias0[r], 0.f);
            o1[r] = (_Float16)fmaxf(a1[r] + bias1[r], 0.f);
        }
        *(v8h*)&y[ob + co0] = o0;
        *(v8h*)&y[ob + co1] = o1;
    }
}

// ===========================================================================
// 2x2 max pool, stride 2, f16 NHWC, 8 channels per thread, shift/mask only.
// ===========================================================================
__global__ void maxpool2x2_nhwc(const _Float16* __restrict__ in,
                                _Float16* __restrict__ out,
                                int C, int W, int cvShift, int woShift,
                                int total_v)
{
    int i = blockIdx.x * blockDim.x + threadIdx.x;
    if (i >= total_v) return;
    const int Wo = 1 << woShift;
    int cv = i & ((1 << cvShift) - 1); int t = i >> cvShift;
    int wo = t & (Wo - 1);             t >>= woShift;
    int ho = t & (Wo - 1);             int bn = t >> woShift;  // H == W

    const int base = (((bn << woShift) + ho) * 2 * W + wo * 2) * C + cv * 8;
    const v8h p00 = *(const v8h*)&in[base];
    const v8h p01 = *(const v8h*)&in[base + C];
    const v8h p10 = *(const v8h*)&in[base + W * C];
    const v8h p11 = *(const v8h*)&in[base + W * C + C];
    v8h m;
    #pragma unroll
    for (int j = 0; j < 8; ++j) {
        _Float16 u = p00[j] > p01[j] ? p00[j] : p01[j];
        _Float16 v = p10[j] > p11[j] ? p10[j] : p11[j];
        m[j] = u > v ? u : v;
    }
    *(v8h*)&out[(((bn << woShift) + ho) << woShift | wo) * C + cv * 8] = m;
}

// ===========================================================================
// Per-model linear from packed f16 weights [b][dout][din]; x f16 [b][32][din].
// One wave per 16x16 tile. fp32_out selects f32 (final, ragged dout=10) vs f16.
// ===========================================================================
__global__ __launch_bounds__(32)
void linear_wmma(const _Float16* __restrict__ x, const _Float16* __restrict__ wp,
                 const float* __restrict__ bias, void* __restrict__ yv,
                 int din, int dout, int fp32_out)
{
    const int b    = blockIdx.y;
    const int m0   = (blockIdx.x >> 1) * 16;
    const int n0   = (blockIdx.x & 1) * 16;
    const int lane = threadIdx.x & 31;
    const int row  = lane & 15;
    const int kb   = (lane < 16) ? 0 : 8;
    const int m    = m0 + row;
    const int mc   = (m < dout) ? m : (dout - 1);   // clamped row (unused rows
                                                    // computed, never stored)
    const _Float16* xb = x  + (b * IMG_BS + n0 + row) * din;
    const _Float16* wr = wp + (b * dout + mc) * din;

    v8f acc = {};
    for (int k0 = 0; k0 < din; k0 += 32) {
        const v8h a0 = *(const v8h*)&wr[k0 + kb];
        const v8h a1 = *(const v8h*)&wr[k0 + kb + 16];
        const v8h b0 = *(const v8h*)&xb[k0 + kb];
        const v8h b1 = *(const v8h*)&xb[k0 + kb + 16];
        v16h av, bv;
        #pragma unroll
        for (int i = 0; i < 8; ++i) {
            av[i] = a0[i]; av[8 + i] = a1[i];
            bv[i] = b0[i]; bv[8 + i] = b1[i];
        }
        acc = __builtin_amdgcn_wmma_f32_16x16x32_f16(
                  false, av, false, bv, (short)0, acc, false, false);
    }

    const int mofs  = (lane < 16) ? 0 : 8;
    const int n_col = n0 + (lane & 15);
    if (fp32_out) {
        float* y = (float*)yv;
        #pragma unroll
        for (int r = 0; r < 8; ++r) {
            int co = m0 + r + mofs;
            if (co < dout)
                y[(b * IMG_BS + n_col) * dout + co] = acc[r] + bias[b * dout + co];
        }
    } else {
        _Float16* y = (_Float16*)yv;
        const int co0 = m0 + mofs;
        const v8f bv8 = *(const v8f*)&bias[b * dout + co0];
        v8h o;
        #pragma unroll
        for (int r = 0; r < 8; ++r) o[r] = (_Float16)(acc[r] + bv8[r]);
        *(v8h*)&y[(b * IMG_BS + n_col) * dout + co0] = o;
    }
}

// ===========================================================================
static inline int ilog2(int v) { int s = 0; while ((1 << s) < v) ++s; return s; }

extern "C" void kernel_launch(void* const* d_in, const int* in_sizes, int n_in,
                              void* d_out, int out_size, void* d_ws, size_t ws_size,
                              hipStream_t stream)
{
    (void)in_sizes; (void)n_in; (void)out_size; (void)ws_size;

    const float* x0 = (const float*)d_in[0];
    const float* Wt[16];
    const float* Bv[16];
    for (int i = 0; i < 16; ++i) {
        Wt[i] = (const float*)d_in[1 + i];
        Bv[i] = (const float*)d_in[17 + i];
    }

    // workspace carve-up (f16): two 8M-half activation buffers + weight pack
    _Float16* h0  = (_Float16*)d_ws;
    _Float16* h1  = h0 + (size_t)8 * 1024 * 1024;
    _Float16* wpk = h1 + (size_t)8 * 1024 * 1024;

    // ---- pack input: f32 NCHW -> f16 NHWC, channels padded 3 -> 32 --------
    {
        int total = IMG_BS * 1024 * 32;
        pack_input<<<dim3((unsigned)((total + 255) / 256)), dim3(256), 0, stream>>>(
            x0, h0, total);
    }

    static const int cfg[18] = {64, 64, -1, 128, 128, -1, 256, 256, 256, -1,
                                512, 512, 512, -1, 512, 512, 512, -1};

    const _Float16* cur = h0;
    _Float16* nxt = h1;
    int C = 3, Cpad = 32, Hc = 32;
    int j = 0;
    for (int t = 0; t < 18; ++t) {
        if (cfg[t] < 0) {
            int Ho = Hc / 2;
            int total_v = MODEL_BS * IMG_BS * Ho * Ho * (C / 8);
            maxpool2x2_nhwc<<<dim3((unsigned)((total_v + 255) / 256)), dim3(256), 0, stream>>>(
                cur, nxt, C, Hc, ilog2(C / 8), ilog2(Ho), total_v);
            Hc >>= 1;
        } else {
            int Cout = cfg[t];
            // pack weights: f32 [b][Cin][Cout][9] -> f16 [b][9][Cout][Cpad]
            int xElems = Cout * Cpad;
            dim3 pgrid((unsigned)((xElems + 255) / 256), 9, MODEL_BS);
            pack_weights<<<pgrid, dim3(256), 0, stream>>>(
                Wt[j], wpk, C, ilog2(Cpad), Cout, 9);
            // conv
            int N  = IMG_BS * Hc * Hc;
            int ms = (j == 0) ? 0 : IMG_BS * Hc * Hc * Cpad;
            dim3 grid((unsigned)(N / 128), (unsigned)(Cout / 64), MODEL_BS);
            conv3x3_wmma<<<grid, dim3(256), 0, stream>>>(
                cur, wpk, Bv[j], nxt, Cpad, Cout, Hc, Hc, ilog2(Hc), ms);
            C = Cout; Cpad = Cout;
            ++j;
        }
        cur = nxt;
        nxt = (nxt == h0) ? h1 : h0;
    }

    // cur: f16 [4][32][512]
    static const int dims[3][2] = {{512, 512}, {512, 512}, {512, 10}};
    for (int li = 0; li < 3; ++li) {
        int din = dims[li][0], dout = dims[li][1];
        // pack linear weights: f32 [b][din][dout] -> f16 [b][dout][din]
        int xElems = dout * din;
        dim3 pgrid((unsigned)((xElems + 255) / 256), 1, MODEL_BS);
        pack_weights<<<pgrid, dim3(256), 0, stream>>>(
            Wt[13 + li], wpk, din, ilog2(din), dout, 1);
        void* outp = (li == 2) ? d_out : (void*)nxt;
        int mtiles = (dout + 15) / 16;
        dim3 grid((unsigned)(mtiles * 2), MODEL_BS);
        linear_wmma<<<grid, dim3(32), 0, stream>>>(
            cur, wpk, Bv[13 + li], outp, din, dout, (li == 2) ? 1 : 0);
        cur = (const _Float16*)outp;
        nxt = (nxt == h0) ? h1 : h0;
    }
}
